// Memory_trans_read_39728447488514
// MI455X (gfx1250) — matrix operand order repaired and checked by reading, not compiled, not run on hardware
//
#include <hip/hip_runtime.h>

typedef __attribute__((ext_vector_type(16))) _Float16 v16h;
typedef __attribute__((ext_vector_type(8)))  _Float16 v8h;
typedef __attribute__((ext_vector_type(8)))  float    v8f;

#define N_Q   65536
#define M_MEM 2048
#define D_DIM 256

// xor-lane swizzle (stays within 16-lane halves for masks < 16), full and_mask
#define SWZ_XOR_F(x, m) \
  __builtin_bit_cast(float, __builtin_amdgcn_ds_swizzle(__builtin_bit_cast(int, (x)), (0x1f | ((m) << 10))))

__device__ __forceinline__ float half_reduce_max(float v) {
  v = fmaxf(v, SWZ_XOR_F(v, 1));
  v = fmaxf(v, SWZ_XOR_F(v, 2));
  v = fmaxf(v, SWZ_XOR_F(v, 4));
  v = fmaxf(v, SWZ_XOR_F(v, 8));
  return v;
}
__device__ __forceinline__ float half_reduce_sum(float v) {
  v += SWZ_XOR_F(v, 1);
  v += SWZ_XOR_F(v, 2);
  v += SWZ_XOR_F(v, 4);
  v += SWZ_XOR_F(v, 8);
  return v;
}

// One async 16B-per-lane copy: global (saddr base + 32-bit voffset) -> LDS (raw lds addr)
__device__ __forceinline__ void async_b128(unsigned lds_addr, unsigned goff,
                                           unsigned long long base) {
  asm volatile("global_load_async_to_lds_b128 %0, %1, %2"
               :: "v"(lds_addr), "v"(goff), "s"(base) : "memory");
}

// memory f32 [2048][256] -> mem_h f16 [2048][256] and memT_h f16 [256][2048]
__global__ __launch_bounds__(256) void cvt_kernel(const float* __restrict__ mem,
                                                  _Float16* __restrict__ mem_h,
                                                  _Float16* __restrict__ memT_h) {
  int idx = blockIdx.x * 256 + threadIdx.x;        // 0 .. 2048*256-1
  float f = mem[idx];
  _Float16 hv = (_Float16)f;
  mem_h[idx] = hv;
  int r = idx >> 8;        // row in [0,2048)
  int c = idx & 255;       // col in [0,256)
  memT_h[c * M_MEM + r] = hv;
}

// Flash-attention style. Each wave handles 32 query rows (2 A-tiles) so every
// K/V B-fragment read from LDS feeds TWO WMMAs (halves DS bytes per WMMA).
// K/V chunks (32 mem rows) staged global->LDS with async copies, double buffered.
__global__ __launch_bounds__(256, 1) void attn_kernel(const float* __restrict__ q,
                                                      const _Float16* __restrict__ kh,   // [2048][256] f16
                                                      const _Float16* __restrict__ vt,   // [256][2048] f16 (transposed)
                                                      float* __restrict__ out) {
  // K chunk staged as [32 mrows][256 d], V chunk staged as [256 d][32 mrows]
  __shared__ __attribute__((aligned(32))) _Float16 kbuf[2][32 * 256];      // 2 x 16 KB
  __shared__ __attribute__((aligned(32))) _Float16 vbuf[2][256 * 32];      // 2 x 16 KB
  __shared__ __attribute__((aligned(32))) _Float16 p_lds[8][2][16][32];    // 16 KB

  const int tid  = threadIdx.x;
  const int lane = tid & 31;
  const int wave = tid >> 5;
  const int n    = lane & 15;   // tile column owned by this lane (B/C layout)
  const int h    = lane >> 4;   // lane half
  const int q0   = blockIdx.x * 256 + wave * 32;

  const unsigned long long khb = (unsigned long long)kh;
  const unsigned long long vtb = (unsigned long long)vt;

  // ---- Load Q (2 tiles x 16 rows x 256) as f16 A-fragments, converting f32->f16 ----
  // A 16x32 layout: lane m=L&15, halves 0..15 = K(h*16 + 0..15)
  v16h qa[2][8];
  #pragma unroll
  for (int qt = 0; qt < 2; ++qt) {
    const float* __restrict__ qrow = q + (size_t)(q0 + qt * 16 + n) * D_DIM;
    #pragma unroll
    for (int kf = 0; kf < 8; ++kf) {
      const float* p0 = qrow + kf * 32 + h * 16;
      #pragma unroll
      for (int i = 0; i < 8; ++i) {
        qa[qt][kf][i]     = (_Float16)p0[i];
        qa[qt][kf][8 + i] = (_Float16)p0[8 + i];
      }
    }
  }

  // ---- Accumulators and online-softmax row state ----
  v8f acc[2][16];
  #pragma unroll
  for (int qt = 0; qt < 2; ++qt)
    #pragma unroll
    for (int t = 0; t < 16; ++t) acc[qt][t] = (v8f){};
  float mrow[2][8], lrow[2][8];
  #pragma unroll
  for (int qt = 0; qt < 2; ++qt)
    #pragma unroll
    for (int v = 0; v < 8; ++v) { mrow[qt][v] = -1.0e30f; lrow[qt][v] = 0.0f; }

  // ---- Stage helper: 256 threads move 16KB K + 16KB V per chunk (8 async ops/wave) ----
  auto stage = [&](int m0, int b) {
    unsigned lk = (unsigned)(uintptr_t)&kbuf[b][0] + (unsigned)tid * 64u;
    unsigned gk = (unsigned)m0 * 512u + (unsigned)tid * 64u;
    #pragma unroll
    for (int j = 0; j < 4; ++j) async_b128(lk + j * 16u, gk + j * 16u, khb);
    unsigned lv = (unsigned)(uintptr_t)&vbuf[b][0] + (unsigned)tid * 64u;
    unsigned gv = (unsigned)tid * 4096u + (unsigned)m0 * 2u;
    #pragma unroll
    for (int j = 0; j < 4; ++j) async_b128(lv + j * 16u, gv + j * 16u, vtb);
  };

  // Prologue: prefetch chunk 0 into buffer 0
  stage(0, 0);

  // ---- Main loop over memory rows in chunks of 32, double buffered ----
  #pragma unroll 1
  for (int c = 0; c < M_MEM / 32; ++c) {
    const int m0  = c * 32;
    const int cur = c & 1;

    // WAR guard: everyone finished reading buffer (c+1)&1 during iteration c-1
    __syncthreads();
    if (c + 1 < M_MEM / 32) {
      stage(m0 + 32, cur ^ 1);
      asm volatile("s_wait_asynccnt 0x8" ::: "memory");   // chunk c's 8 copies done
    } else {
      asm volatile("s_wait_asynccnt 0x0" ::: "memory");
    }
    __syncthreads();                                      // all waves' chunk-c data in LDS

    const _Float16* kcur = &kbuf[cur][0];
    const _Float16* vcur = &vbuf[cur][0];

    // S tiles: s[qt][t] = Q_qt @ K[m0+t*16 ..]^T  (each B fragment feeds 2 WMMAs)
    v8f s[2][2];
    #pragma unroll
    for (int qt = 0; qt < 2; ++qt) { s[qt][0] = (v8f){}; s[qt][1] = (v8f){}; }
    #pragma unroll
    for (int kf = 0; kf < 8; ++kf) {
      v16h b0 = *(const v16h*)&kcur[(size_t)n * 256 + kf * 32 + h * 16];
      v16h b1 = *(const v16h*)&kcur[(size_t)(16 + n) * 256 + kf * 32 + h * 16];
      #pragma unroll
      for (int qt = 0; qt < 2; ++qt) {
        s[qt][0] = __builtin_amdgcn_wmma_f32_16x16x32_f16(false, qa[qt][kf], false, b0, (short)0, s[qt][0], false, false);
        s[qt][1] = __builtin_amdgcn_wmma_f32_16x16x32_f16(false, qa[qt][kf], false, b1, (short)0, s[qt][1], false, false);
      }
    }

    // ---- Online softmax update (row r = v + 8*h, column n), per query tile ----
    #pragma unroll
    for (int qt = 0; qt < 2; ++qt) {
      float scale[8];
      #pragma unroll
      for (int v = 0; v < 8; ++v) {
        float rm   = half_reduce_max(fmaxf(s[qt][0][v], s[qt][1][v]));
        float mnew = fmaxf(mrow[qt][v], rm);
        float sc   = __expf(mrow[qt][v] - mnew);
        float p0   = __expf(s[qt][0][v] - mnew);
        float p1   = __expf(s[qt][1][v] - mnew);
        p_lds[wave][qt][v + 8 * h][n]      = (_Float16)p0;
        p_lds[wave][qt][v + 8 * h][16 + n] = (_Float16)p1;
        lrow[qt][v] = lrow[qt][v] * sc + half_reduce_sum(p0 + p1);
        mrow[qt][v] = mnew;
        scale[v] = sc;
      }
      #pragma unroll
      for (int t = 0; t < 16; ++t) {
        #pragma unroll
        for (int v = 0; v < 8; ++v) acc[qt][t][v] *= scale[v];
      }
    }

    // P tiles written by this wave only -> wait for our DS stores
    asm volatile("s_wait_dscnt 0" ::: "memory");

    // Reload P as A-fragments (C-layout -> A-layout transpose via LDS)
    v16h pa[2];
    #pragma unroll
    for (int qt = 0; qt < 2; ++qt) {
      const _Float16* pl = &p_lds[wave][qt][n][h * 16];
      v8h pa0 = *(const v8h*)(pl);
      v8h pa1 = *(const v8h*)(pl + 8);
      #pragma unroll
      for (int i = 0; i < 8; ++i) { pa[qt][i] = pa0[i]; pa[qt][8 + i] = pa1[i]; }
    }

    // O(32x256) += P(32x32) @ V(32x256); each V B-fragment feeds 2 WMMAs
    #pragma unroll
    for (int td = 0; td < 16; ++td) {
      v16h b = *(const v16h*)&vcur[(size_t)(td * 16 + n) * 32 + h * 16];
      acc[0][td] = __builtin_amdgcn_wmma_f32_16x16x32_f16(false, pa[0], false, b, (short)0, acc[0][td], false, false);
      acc[1][td] = __builtin_amdgcn_wmma_f32_16x16x32_f16(false, pa[1], false, b, (short)0, acc[1][td], false, false);
    }
  }

  // ---- Epilogue: O /= l, store f32 ----
  #pragma unroll
  for (int qt = 0; qt < 2; ++qt) {
    #pragma unroll
    for (int v = 0; v < 8; ++v) lrow[qt][v] = 1.0f / lrow[qt][v];
    #pragma unroll
    for (int td = 0; td < 16; ++td) {
      #pragma unroll
      for (int v = 0; v < 8; ++v) {
        out[(size_t)(q0 + qt * 16 + v + 8 * h) * D_DIM + td * 16 + n] = acc[qt][td][v] * lrow[qt][v];
      }
    }
  }
}

extern "C" void kernel_launch(void* const* d_in, const int* in_sizes, int n_in,
                              void* d_out, int out_size, void* d_ws, size_t ws_size,
                              hipStream_t stream) {
  const float* memory = (const float*)d_in[0];   // [2048][256] f32
  const float* query  = (const float*)d_in[1];   // [65536][256] f32
  float* out = (float*)d_out;                    // [65536][256] f32

  _Float16* mem_h  = (_Float16*)d_ws;                                          // 1 MB
  _Float16* memT_h = (_Float16*)((char*)d_ws + (size_t)M_MEM * D_DIM * 2);     // 1 MB

  cvt_kernel<<<(M_MEM * D_DIM) / 256, 256, 0, stream>>>(memory, mem_h, memT_h);
  attn_kernel<<<N_Q / 256, 256, 0, stream>>>(query, mem_h, memT_h, out);
}
